// Head_64527588655108
// MI455X (gfx1250) — matrix operand order repaired and checked by reading, not compiled, not run on hardware
//
#include <hip/hip_runtime.h>
#include <hip/hip_bf16.h>

// ---------------------------------------------------------------------------
// Causal single-head attention (B=16, T=2048, C=1024, H=64) for gfx1250.
// Pass 1: Wq/Wk/Wv -> f16, transposed (H,C).
// Pass 2: Q = q@Wq, K = k@Wk (B,T,H f16), V = k@Wv stored transposed (B,H,T f16)
//         via v_wmma_f32_16x16x32_f16 (12 WMMAs per 32-wide C step).
// Pass 3: flash attention, one wave per 16-row query tile, 32-wide key tiles,
//         8 WMMAs per tile + online softmax in registers.
// ---------------------------------------------------------------------------

typedef __attribute__((ext_vector_type(16))) _Float16 v16h;
typedef __attribute__((ext_vector_type(8)))  _Float16 v8h;
typedef __attribute__((ext_vector_type(8)))  float    v8f;

namespace cfg {
constexpr int B = 16;
constexpr int T = 2048;
constexpr int C = 1024;
constexpr int H = 64;
constexpr float SCALE = 0.03125f;  // C^-0.5 = 1/32 (reference scales by input dim C)
}

__device__ __forceinline__ v8f wmma_f16(v16h a, v16h b, v8f c) {
  // D = A(16x32) * B(32x16) + C ; f32 accumulate
  return __builtin_amdgcn_wmma_f32_16x16x32_f16(
      /*neg_a=*/false, a, /*neg_b=*/false, b,
      /*c_mod=*/(short)0, c, /*reuse_a=*/false, /*reuse_b=*/false);
}

// A-operand (16x32 f16) gather from a row-major matrix, K-dim contiguous.
// lane g holds row M=g; hi selects K chunk {hi*16..hi*16+15}
__device__ __forceinline__ v16h load_a_f16(const _Float16* rowptr, int kbase, int hi) {
  v8h lo = *(const v8h*)(rowptr + kbase + hi * 16);
  v8h hi8 = *(const v8h*)(rowptr + kbase + hi * 16 + 8);
  v16h r;
#pragma unroll
  for (int i = 0; i < 8; ++i) { r[i] = lo[i]; r[8 + i] = hi8[i]; }
  return r;
}

// Same A-operand pattern but converting from f32 source.
__device__ __forceinline__ v16h load_a_f32(const float* rowptr, int kbase, int hi) {
  v16h r;
#pragma unroll
  for (int i = 0; i < 8; ++i) r[i] = (_Float16)rowptr[kbase + hi * 16 + i];
#pragma unroll
  for (int i = 0; i < 8; ++i) r[8 + i] = (_Float16)rowptr[kbase + hi * 16 + 8 + i];
  return r;
}

__device__ __forceinline__ float redmax16(float x) {
  x = fmaxf(x, __shfl_xor(x, 1, 32));
  x = fmaxf(x, __shfl_xor(x, 2, 32));
  x = fmaxf(x, __shfl_xor(x, 4, 32));
  x = fmaxf(x, __shfl_xor(x, 8, 32));
  return x;  // uniform across each 16-lane half-group
}
__device__ __forceinline__ float redsum16(float x) {
  x += __shfl_xor(x, 1, 32);
  x += __shfl_xor(x, 2, 32);
  x += __shfl_xor(x, 4, 32);
  x += __shfl_xor(x, 8, 32);
  return x;
}

// ---------------------------------------------------------------------------
// Pass 1: weight transpose + f16 convert: Wt[h*C + c] = (f16)W[c*H + h]
// ---------------------------------------------------------------------------
__global__ void wt_kernel(const float* __restrict__ Wq, const float* __restrict__ Wk,
                          const float* __restrict__ Wv,
                          _Float16* __restrict__ Wtq, _Float16* __restrict__ Wtk,
                          _Float16* __restrict__ Wtv) {
  using namespace cfg;
  int idx = blockIdx.x * blockDim.x + threadIdx.x;  // idx = h*C + c
  if (idx >= C * H) return;
  int h = idx / C;
  int c = idx - h * C;
  int src = c * H + h;
  Wtq[idx] = (_Float16)Wq[src];
  Wtk[idx] = (_Float16)Wk[src];
  Wtv[idx] = (_Float16)Wv[src];
}

// ---------------------------------------------------------------------------
// Pass 2: projections. One wave per 16-row tile; 128 threads = 4 waves/block.
//   Qw, Kw : (B,T,H) f16 row-major
//   Vt     : (B,H,T) f16 (transposed so attention loads it as WMMA-B directly)
// ---------------------------------------------------------------------------
__global__ void __launch_bounds__(128) proj_kernel(
    const float* __restrict__ q, const float* __restrict__ k,
    const _Float16* __restrict__ Wtq, const _Float16* __restrict__ Wtk,
    const _Float16* __restrict__ Wtv,
    _Float16* __restrict__ Qw, _Float16* __restrict__ Kw, _Float16* __restrict__ Vt) {
  using namespace cfg;
  int wave = blockIdx.x * (blockDim.x >> 5) + (threadIdx.x >> 5);  // [0, B*T/16)
  int b = wave >> 7;                                               // T/16 = 128
  int t0 = (wave & 127) * 16;
  int lane = threadIdx.x & 31;
  int g = lane & 15, hi = lane >> 4;

  const float* qrow = q + ((size_t)b * T + t0 + g) * C;
  const float* krow = k + ((size_t)b * T + t0 + g) * C;

  v8f accQ[4] = {}, accK[4] = {}, accV[4] = {};
  for (int c0 = 0; c0 < C; c0 += 32) {
    v16h aq = load_a_f32(qrow, c0, hi);
    v16h ak = load_a_f32(krow, c0, hi);
#pragma unroll
    for (int j = 0; j < 4; ++j) {
      size_t wrow = (size_t)(j * 16 + g) * C + c0 + hi * 16;  // B operand: 16 contig halfs
      v16h bq = *(const v16h*)(Wtq + wrow);
      v16h bk = *(const v16h*)(Wtk + wrow);
      v16h bv = *(const v16h*)(Wtv + wrow);
      accQ[j] = wmma_f16(aq, bq, accQ[j]);
      accK[j] = wmma_f16(ak, bk, accK[j]);
      accV[j] = wmma_f16(ak, bv, accV[j]);
    }
  }
  // C/D layout: row = r + 8*hi, col = g (per 16-col tile j)
#pragma unroll
  for (int j = 0; j < 4; ++j) {
    int hcol = j * 16 + g;
#pragma unroll
    for (int r = 0; r < 8; ++r) {
      int trow = t0 + r + 8 * hi;
      Qw[((size_t)b * T + trow) * H + hcol] = (_Float16)accQ[j][r];
      Kw[((size_t)b * T + trow) * H + hcol] = (_Float16)accK[j][r];
    }
    v8h pv;
#pragma unroll
    for (int r = 0; r < 8; ++r) pv[r] = (_Float16)accV[j][r];
    // Vt: column h becomes a row; 8 consecutive t's are contiguous -> b128 store
    *(v8h*)(Vt + ((size_t)b * H + hcol) * T + t0 + 8 * hi) = pv;
  }
}

// ---------------------------------------------------------------------------
// Pass 3: flash attention. One wave owns 16 query rows; loops 32-wide key tiles.
// ---------------------------------------------------------------------------
__global__ void __launch_bounds__(128) attn_kernel(
    const _Float16* __restrict__ Qw, const _Float16* __restrict__ Kw,
    const _Float16* __restrict__ Vt, float* __restrict__ out) {
  using namespace cfg;
  __shared__ __align__(32) _Float16 plds[4][16 * 32];  // per-wave P-tile transpose
  int wslot = threadIdx.x >> 5;
  int wave = blockIdx.x * 4 + wslot;  // [0, B*T/16)
  int b = wave >> 7;
  int t0 = (wave & 127) * 16;
  int lane = threadIdx.x & 31;
  int g = lane & 15, hi = lane >> 4;

  const _Float16* Qb = Qw + (size_t)b * T * H;
  const _Float16* Kb = Kw + (size_t)b * T * H;
  const _Float16* Vb = Vt + (size_t)b * H * T;
  _Float16* pl = &plds[wslot][0];

  // Q tile as A operand, split over h: kbase 0 and 32
  const _Float16* qrow = Qb + (size_t)(t0 + g) * H;
  v16h aq0 = load_a_f16(qrow, 0, hi);
  v16h aq1 = load_a_f16(qrow, 32, hi);

  float mrow[8], lrow[8];
  v8f o[4] = {};
#pragma unroll
  for (int r = 0; r < 8; ++r) { mrow[r] = -3.0e38f; lrow[r] = 0.0f; }

  for (int s0 = 0; s0 < t0 + 16; s0 += 32) {
    // ---- S = Q K^T for key cols [s0, s0+32) : 4 WMMAs ----
    const _Float16* krow0 = Kb + (size_t)(s0 + g) * H;       // cols s0..s0+15
    const _Float16* krow1 = Kb + (size_t)(s0 + 16 + g) * H;  // cols s0+16..s0+31
    v8f sa = {}, sb = {};
    sa = wmma_f16(aq0, *(const v16h*)(krow0 + hi * 16), sa);
    sa = wmma_f16(aq1, *(const v16h*)(krow0 + 32 + hi * 16), sa);
    sb = wmma_f16(aq0, *(const v16h*)(krow1 + hi * 16), sb);
    sb = wmma_f16(aq1, *(const v16h*)(krow1 + 32 + hi * 16), sb);

    if (s0 + 32 < t0 + 16) {  // prefetch next key/value tile -> global_prefetch_b8
      __builtin_prefetch(Kb + (size_t)(s0 + 32 + g) * H, 0, 0);
      __builtin_prefetch(Vb + (size_t)g * T + s0 + 32, 0, 0);
    }

    // ---- online softmax (per row r+8*hi, cols spread over 16-lane group) ----
    float p0[8], p1[8], fr[8];
#pragma unroll
    for (int r = 0; r < 8; ++r) {
      int trow = t0 + r + 8 * hi;
      float x0 = sa[r] * SCALE;
      float x1 = sb[r] * SCALE;
      if (s0 + g > trow) x0 = -1.0e30f;        // causal mask
      if (s0 + 16 + g > trow) x1 = -1.0e30f;
      float tmax = redmax16(fmaxf(x0, x1));
      float mnew = fmaxf(mrow[r], tmax);
      float f = __expf(mrow[r] - mnew);
      mrow[r] = mnew;
      float e0 = __expf(x0 - mnew);
      float e1 = __expf(x1 - mnew);
      lrow[r] = lrow[r] * f + redsum16(e0 + e1);
      fr[r] = f;
      p0[r] = e0;
      p1[r] = e1;
    }
#pragma unroll
    for (int j = 0; j < 4; ++j)
#pragma unroll
      for (int r = 0; r < 8; ++r) o[j][r] *= fr[r];

    // ---- P (C layout) -> LDS row-major 16x32, reload in A layout ----
#pragma unroll
    for (int r = 0; r < 8; ++r) {
      int row = r + 8 * hi;
      pl[row * 32 + g] = (_Float16)p0[r];
      pl[row * 32 + 16 + g] = (_Float16)p1[r];
    }
    asm volatile("s_wait_dscnt 0" ::: "memory");  // wave-local store->load fence
    v16h pa = load_a_f16(pl + g * 32, 0, hi);

    // ---- O += P @ V : 4 WMMAs, V already in B layout via Vt (H,T) ----
#pragma unroll
    for (int j = 0; j < 4; ++j) {
      v16h bv = *(const v16h*)(Vb + (size_t)(j * 16 + g) * T + s0 + hi * 16);
      o[j] = wmma_f16(pa, bv, o[j]);
    }
  }

  // ---- epilogue: out = O / l ----
#pragma unroll
  for (int j = 0; j < 4; ++j) {
    int hcol = j * 16 + g;
#pragma unroll
    for (int r = 0; r < 8; ++r) {
      int trow = t0 + r + 8 * hi;
      out[((size_t)b * T + trow) * H + hcol] = o[j][r] / lrow[r];
    }
  }
}

// ---------------------------------------------------------------------------
extern "C" void kernel_launch(void* const* d_in, const int* in_sizes, int n_in,
                              void* d_out, int out_size, void* d_ws, size_t ws_size,
                              hipStream_t stream) {
  using namespace cfg;
  (void)in_sizes; (void)n_in; (void)out_size; (void)ws_size;
  const float* q  = (const float*)d_in[0];
  const float* k  = (const float*)d_in[1];
  const float* Wq = (const float*)d_in[2];
  const float* Wk = (const float*)d_in[3];
  const float* Wv = (const float*)d_in[4];
  float* out = (float*)d_out;

  // f16 workspace layout (halfs): Wt (3 * H*C) then Qw, Kw, Vt (each B*T*H)
  _Float16* ws  = (_Float16*)d_ws;
  _Float16* Wtq = ws;
  _Float16* Wtk = Wtq + (size_t)H * C;
  _Float16* Wtv = Wtk + (size_t)H * C;
  _Float16* Qw  = Wtv + (size_t)H * C;
  _Float16* Kw  = Qw + (size_t)B * T * H;
  _Float16* Vt  = Kw + (size_t)B * T * H;

  wt_kernel<<<(C * H + 255) / 256, 256, 0, stream>>>(Wq, Wk, Wv, Wtq, Wtk, Wtv);

  int tiles = B * (T / 16);  // 2048 wave-tiles, 4 waves per 128-thread block
  proj_kernel<<<tiles / 4, 128, 0, stream>>>(q, k, Wtq, Wtk, Wtv, Qw, Kw, Vt);
  attn_kernel<<<tiles / 4, 128, 0, stream>>>(Qw, Kw, Vt, out);
}
